// RoPE_30313879175566
// MI455X (gfx1250) — matrix operand order repaired and checked by reading, not compiled, not run on hardware
//
#include <hip/hip_runtime.h>

// ---------------------------------------------------------------------------
// Types for CDNA5 WMMA (wave32, 16x16x32 bf16 -> f32 accumulate)
// ---------------------------------------------------------------------------
typedef __attribute__((ext_vector_type(16))) __bf16 v16bf;
typedef __attribute__((ext_vector_type(8)))  float  v8f;

union FragB {
    v16bf        v;
    unsigned int u[8];
};

__device__ __forceinline__ v8f wmma_bf16(const FragB& a, const FragB& b, v8f c) {
    return __builtin_amdgcn_wmma_f32_16x16x32_bf16(
        /*neg_a=*/false, a.v, /*neg_b=*/false, b.v,
        /*c_mod=*/(short)0, c, /*reuse_a=*/false, /*reuse_b=*/false);
}

// round-to-nearest-even f32 -> bf16 (stored as raw u16)
__device__ __forceinline__ unsigned short f2b(float x) {
    unsigned int u = __builtin_bit_cast(unsigned int, x);
    unsigned int r = u + 0x7FFFu + ((u >> 16) & 1u);
    return (unsigned short)(r >> 16);
}

// Problem constants
#define BATCH 2
#define SEQ   2048
#define DMODEL 1024
#define NHEAD 16
#define HD    64

// ---------------------------------------------------------------------------
// Elementwise f32 -> bf16 cast
// ---------------------------------------------------------------------------
__global__ __launch_bounds__(256) void cast_f32_bf16(const float* __restrict__ in,
                                                     unsigned short* __restrict__ out,
                                                     int n) {
    int i = blockIdx.x * blockDim.x + threadIdx.x;
    if (i < n) out[i] = f2b(in[i]);
}

// ---------------------------------------------------------------------------
// bf16 GEMM:  C[M,N] = A[M,K] * B[N,K]^T + bias[N]    (A,B bf16, C f32)
// Block tile 128x128, K-step 32, 8 waves (2 along M x 4 along N),
// wave tile 64x32 = 4x2 frags of 16x16 -> 8 WMMAs per K-step.
// ---------------------------------------------------------------------------
#define BM 128
#define BN 128
#define BK 32

__global__ __launch_bounds__(256) void gemm_bf16_nt(const unsigned short* __restrict__ A,
                                                    const unsigned short* __restrict__ B,
                                                    const float* __restrict__ bias,
                                                    float* __restrict__ C,
                                                    int M, int N, int K) {
    __shared__ unsigned short sA[BM][BK + 2];  // [m][k]
    __shared__ unsigned short sB[BK][BN + 2];  // [k][n]  (transposed on store)

    const int t    = threadIdx.x;
    const int w    = t >> 5;
    const int lane = t & 31;
    const int lm   = lane & 15;
    const int hi   = lane >> 4;

    const int bm = blockIdx.y * BM;
    const int bn = blockIdx.x * BN;
    const int wm = (w & 1) * 64;   // 2 waves along M, wave covers 64 rows
    const int wn = (w >> 1) * 32;  // 4 waves along N, wave covers 32 cols

    v8f acc[4][2] = {};

    const int srow = t >> 1, scol = (t & 1) * 16; // staging mapping (both tiles)

    for (int k0 = 0; k0 < K; k0 += BK) {
        // ---- stage A tile (128x32): 16 bf16 per thread ----
        {
            const uint4* gp = (const uint4*)(A + (size_t)(bm + srow) * K + k0 + scol);
            uint4 d0 = gp[0], d1 = gp[1];
            unsigned int* sp = (unsigned int*)&sA[srow][scol];
            sp[0] = d0.x; sp[1] = d0.y; sp[2] = d0.z; sp[3] = d0.w;
            sp[4] = d1.x; sp[5] = d1.y; sp[6] = d1.z; sp[7] = d1.w;
        }
        // ---- stage B tile (128 n-rows x 32 k) transposed into sB[k][n] ----
        {
            const uint4* gp = (const uint4*)(B + (size_t)(bn + srow) * K + k0 + scol);
            uint4 d0 = gp[0], d1 = gp[1];
            unsigned int dv[8] = {d0.x, d0.y, d0.z, d0.w, d1.x, d1.y, d1.z, d1.w};
#pragma unroll
            for (int i = 0; i < 8; ++i) {
                sB[scol + 2 * i][srow]     = (unsigned short)(dv[i] & 0xFFFFu);
                sB[scol + 2 * i + 1][srow] = (unsigned short)(dv[i] >> 16);
            }
        }
        if (k0 + BK < K) {
            __builtin_prefetch(A + (size_t)(bm + srow) * K + k0 + BK + scol, 0, 3);
            __builtin_prefetch(B + (size_t)(bn + srow) * K + k0 + BK + scol, 0, 3);
        }
        __syncthreads();

        // ---- fragments ----
        FragB aF[4], bF[2];
#pragma unroll
        for (int fm = 0; fm < 4; ++fm) {
            const int m = wm + fm * 16 + lm;
#pragma unroll
            for (int j = 0; j < 8; ++j) {
                const int kb = ((j & 4) ? 16 : 0) + hi * 8 + (j & 3) * 2;
                aF[fm].u[j] = *(const unsigned int*)&sA[m][kb];
            }
        }
#pragma unroll
        for (int fn = 0; fn < 2; ++fn) {
            const int n = wn + fn * 16 + lm;
#pragma unroll
            for (int j = 0; j < 8; ++j) {
                const int kb = ((j & 4) ? 16 : 0) + hi * 8 + (j & 3) * 2;
                bF[fn].u[j] = (unsigned int)sB[kb][n] | ((unsigned int)sB[kb + 1][n] << 16);
            }
        }
#pragma unroll
        for (int fm = 0; fm < 4; ++fm)
#pragma unroll
            for (int fn = 0; fn < 2; ++fn)
                acc[fm][fn] = wmma_bf16(aF[fm], bF[fn], acc[fm][fn]);
        __syncthreads();
    }

    // ---- epilogue: C/D layout N=lane&15, M=r+8*hi ----
#pragma unroll
    for (int fm = 0; fm < 4; ++fm)
#pragma unroll
        for (int fn = 0; fn < 2; ++fn) {
#pragma unroll
            for (int r = 0; r < 8; ++r) {
                const int row = bm + wm + fm * 16 + r + 8 * hi;
                const int col = bn + wn + fn * 16 + lm;
                C[(size_t)row * N + col] = acc[fm][fn][r] + bias[col];
            }
        }
}

// ---------------------------------------------------------------------------
// RoPE + head split.  qkv: f32 [B, L, 3*D].  One thread per (b,h,l,pair j<32).
// Writes bf16 q/k/v in [B,H,L,64]; f32 roped-k and v into d_out regions.
// ---------------------------------------------------------------------------
__global__ __launch_bounds__(256) void rope_split(const float* __restrict__ qkv,
                                                  unsigned short* __restrict__ qb,
                                                  unsigned short* __restrict__ kb,
                                                  unsigned short* __restrict__ vb,
                                                  float* __restrict__ k_out,
                                                  float* __restrict__ v_out) {
    const int idx = blockIdx.x * blockDim.x + threadIdx.x; // B*H*L*32 threads
    const int j = idx & 31;
    const int l = (idx >> 5) & (SEQ - 1);
    const int h = (idx >> 16) & (NHEAD - 1);
    const int b = idx >> 20;

    const float inv_freq = __expf(-(float)j * (2.0f / 64.0f) * 9.210340371976184f); // ln(10000)
    const float ang = (float)l * inv_freq;
    const float c = __cosf(ang);
    const float s = __sinf(ang);

    const size_t base = ((size_t)b * SEQ + l) * (3 * DMODEL) + h * HD;
    const float q1 = qkv[base + j],              q2 = qkv[base + j + 32];
    const float k1 = qkv[base + DMODEL + j],     k2 = qkv[base + DMODEL + j + 32];
    const float v1 = qkv[base + 2 * DMODEL + j], v2 = qkv[base + 2 * DMODEL + j + 32];

    const float qr1 = q1 * c - q2 * s, qr2 = q1 * s + q2 * c;
    const float kr1 = k1 * c - k2 * s, kr2 = k1 * s + k2 * c;

    const size_t ho = (((size_t)b * NHEAD + h) * SEQ + l) * HD;
    qb[ho + j] = f2b(qr1); qb[ho + j + 32] = f2b(qr2);
    kb[ho + j] = f2b(kr1); kb[ho + j + 32] = f2b(kr2);
    vb[ho + j] = f2b(v1);  vb[ho + j + 32] = f2b(v2);
    k_out[ho + j] = kr1; k_out[ho + j + 32] = kr2;
    v_out[ho + j] = v1;  v_out[ho + j + 32] = v2;
}

// ---------------------------------------------------------------------------
// Flash attention, causal.  grid = (B*H, L/128).  8 waves, 16 q-rows per wave.
// q/k/v bf16 in [B,H,L,64].  attn out f32 [B,L,D].
// V tile staged via GLOBAL_LOAD_ASYNC_TO_LDS_B128 (ASYNCcnt path).
// ---------------------------------------------------------------------------
__global__ __launch_bounds__(256) void attn_fa(const unsigned short* __restrict__ qb,
                                               const unsigned short* __restrict__ kbuf,
                                               const unsigned short* __restrict__ vbuf,
                                               float* __restrict__ attn) {
    __shared__ unsigned short Kt[HD][66];          // [d][key_local]  (transposed)
    __shared__ unsigned short Vt[64][72];          // [key_local][d], 144B rows (16B aligned)
    __shared__ unsigned short Pl[8][16 * 34];      // per-wave 16x32 P tile

    const int bh   = blockIdx.x;          // 0..31
    const int qblk = blockIdx.y;          // 0..15
    const int b = bh >> 4, h = bh & 15;

    const int t    = threadIdx.x;
    const int w    = t >> 5;
    const int lane = t & 31;
    const int lm   = lane & 15;
    const int hi   = lane >> 4;
    const int q0   = qblk * 128 + w * 16;

    const unsigned short* qh = qb   + (size_t)bh * SEQ * HD;
    const unsigned short* kh = kbuf + (size_t)bh * SEQ * HD;
    const unsigned short* vh = vbuf + (size_t)bh * SEQ * HD;

    // Q fragments: A-layout 16x32, two K-chunks covering d=0..63
    FragB qF[2];
    {
        const int m = q0 + lm;
#pragma unroll
        for (int c2 = 0; c2 < 2; ++c2)
#pragma unroll
            for (int j = 0; j < 8; ++j) {
                const int kd = c2 * 32 + ((j & 4) ? 16 : 0) + hi * 8 + (j & 3) * 2;
                qF[c2].u[j] = *(const unsigned int*)&qh[(size_t)m * HD + kd];
            }
    }

    v8f o[4] = {};
    float rmax[8], rsum[8];
#pragma unroll
    for (int r = 0; r < 8; ++r) { rmax[r] = -3.0e38f; rsum[r] = 0.0f; }

    const int kend = qblk * 128 + 128; // causal horizon for this block
    const int skey = t >> 2, sd = (t & 3) * 16;

    for (int kb0 = 0; kb0 < kend; kb0 += 64) {
        // ---- async-stage V tile (64 keys x 64 dims): 2x b128 per thread ----
        {
            const unsigned short* src = &vh[(size_t)(kb0 + skey) * HD + sd];
            unsigned int       lds0 = (unsigned int)(size_t)&Vt[skey][sd];
            unsigned long long ga0  = (unsigned long long)(size_t)src;
            asm volatile("global_load_async_to_lds_b128 %0, %1, off"
                         :: "v"(lds0), "v"(ga0) : "memory");
            unsigned int       lds1 = lds0 + 16;
            unsigned long long ga1  = ga0 + 16;
            asm volatile("global_load_async_to_lds_b128 %0, %1, off"
                         :: "v"(lds1), "v"(ga1) : "memory");
        }
        // ---- stage K tile transposed: Kt[d][key] ----
        {
            const uint4* kp = (const uint4*)&kh[(size_t)(kb0 + skey) * HD + sd];
            uint4 a0 = kp[0], a1 = kp[1];
            unsigned int av[8] = {a0.x, a0.y, a0.z, a0.w, a1.x, a1.y, a1.z, a1.w};
#pragma unroll
            for (int i = 0; i < 8; ++i) {
                Kt[sd + 2 * i][skey]     = (unsigned short)(av[i] & 0xFFFFu);
                Kt[sd + 2 * i + 1][skey] = (unsigned short)(av[i] >> 16);
            }
        }
        asm volatile("s_wait_asynccnt 0x0" ::: "memory");
        __syncthreads();

#pragma unroll
        for (int skb = 0; skb < 2; ++skb) {
            const int kbase = kb0 + skb * 32;
            if (kbase > q0 + 15) break; // fully masked for this wave

            // ---- S = Q K^T over 32 keys (two 16-key column fragments) ----
            v8f s0 = {}, s1 = {};
#pragma unroll
            for (int c2 = 0; c2 < 2; ++c2) {
                FragB k0f, k1f;
                const int nl = skb * 32 + lm;
#pragma unroll
                for (int j = 0; j < 8; ++j) {
                    const int dk = c2 * 32 + ((j & 4) ? 16 : 0) + hi * 8 + (j & 3) * 2;
                    k0f.u[j] = (unsigned int)Kt[dk][nl] | ((unsigned int)Kt[dk + 1][nl] << 16);
                    k1f.u[j] = (unsigned int)Kt[dk][nl + 16] | ((unsigned int)Kt[dk + 1][nl + 16] << 16);
                }
                s0 = wmma_bf16(qF[c2], k0f, s0);
                s1 = wmma_bf16(qF[c2], k1f, s1);
            }

            // ---- online softmax (per-row over 16 lanes of each half) ----
            const float scale = 0.125f; // 1/sqrt(64)
            float p0[8], p1[8];
#pragma unroll
            for (int r = 0; r < 8; ++r) {
                const int row = q0 + 8 * hi + r;
                float v0 = s0[r] * scale + ((kbase + lm)      > row ? -1.0e9f : 0.0f);
                float v1 = s1[r] * scale + ((kbase + 16 + lm) > row ? -1.0e9f : 0.0f);
                float mx = fmaxf(v0, v1);
#pragma unroll
                for (int off = 1; off < 16; off <<= 1)
                    mx = fmaxf(mx, __shfl_xor(mx, off, 32));
                const float newm  = fmaxf(rmax[r], mx);
                const float alpha = __expf(rmax[r] - newm);
                const float e0 = __expf(v0 - newm);
                const float e1 = __expf(v1 - newm);
                float sum = e0 + e1;
#pragma unroll
                for (int off = 1; off < 16; off <<= 1)
                    sum += __shfl_xor(sum, off, 32);
                rsum[r] = rsum[r] * alpha + sum;
                rmax[r] = newm;
                p0[r] = e0; p1[r] = e1;
#pragma unroll
                for (int dc = 0; dc < 4; ++dc) o[dc][r] *= alpha;
            }

            // ---- redistribute P (C-layout) -> A-layout via per-wave LDS ----
            unsigned short* pw = &Pl[w][0];
#pragma unroll
            for (int r = 0; r < 8; ++r) {
                const int mrow = r + 8 * hi;
                pw[mrow * 34 + lm]      = f2b(p0[r]);
                pw[mrow * 34 + 16 + lm] = f2b(p1[r]);
            }
            FragB pF;
#pragma unroll
            for (int j = 0; j < 8; ++j) {
                const int kp = ((j & 4) ? 16 : 0) + hi * 8 + (j & 3) * 2;
                pF.u[j] = *(const unsigned int*)&pw[lm * 34 + kp];
            }

            // ---- O += P V ----
#pragma unroll
            for (int dc = 0; dc < 4; ++dc) {
                FragB vF;
#pragma unroll
                for (int j = 0; j < 8; ++j) {
                    const int kk2 = skb * 32 + ((j & 4) ? 16 : 0) + hi * 8 + (j & 3) * 2;
                    vF.u[j] = (unsigned int)Vt[kk2][dc * 16 + lm] | ((unsigned int)Vt[kk2 + 1][dc * 16 + lm] << 16);
                }
                o[dc] = wmma_bf16(pF, vF, o[dc]);
            }
        }
        __syncthreads();
    }

    // ---- normalize + write attn [B, L, D] ----
#pragma unroll
    for (int dc = 0; dc < 4; ++dc)
#pragma unroll
        for (int r = 0; r < 8; ++r) {
            const int row = q0 + r + 8 * hi;
            const float val = o[dc][r] / rsum[r];
            attn[(((size_t)b * SEQ + row) * DMODEL) + h * HD + dc * 16 + lm] = val;
        }
}

// ---------------------------------------------------------------------------
// Launch
// ---------------------------------------------------------------------------
extern "C" void kernel_launch(void* const* d_in, const int* in_sizes, int n_in,
                              void* d_out, int out_size, void* d_ws, size_t ws_size,
                              hipStream_t stream) {
    (void)in_sizes; (void)n_in; (void)out_size; (void)ws_size;

    const float* x       = (const float*)d_in[0];
    /* d_in[1] = mask: causal mask applied analytically */
    const float* Wqkv_w  = (const float*)d_in[2];
    const float* Wqkv_b  = (const float*)d_in[3];
    const float* Wout_w  = (const float*)d_in[4];
    const float* Wout_b  = (const float*)d_in[5];

    const int ML  = BATCH * SEQ;              // 4096
    const int NQ  = 3 * DMODEL;               // 3072
    const int HLD = BATCH * NHEAD * SEQ * HD; // 4194304

    // workspace carve-up (bytes)
    char* ws = (char*)d_ws;
    unsigned short* xb    = (unsigned short*)(ws);                    // 8 MiB
    unsigned short* wqkvb = (unsigned short*)(ws + 8388608);          // 6 MiB
    unsigned short* woutb = (unsigned short*)(ws + 14680064);         // 2 MiB
    float*          qkv   = (float*)        (ws + 16777216);          // 48 MiB
    unsigned short* qb    = (unsigned short*)(ws + 67108864);         // 8 MiB
    unsigned short* kb    = (unsigned short*)(ws + 75497472);         // 8 MiB
    unsigned short* vb    = (unsigned short*)(ws + 83886080);         // 8 MiB
    float*          attn  = (float*)        (ws + 92274688);          // 16 MiB
    unsigned short* attnb = (unsigned short*)(ws + 109051904);        // 8 MiB

    float* out_main = (float*)d_out;                  // [B,L,D]
    float* k_out    = out_main + (size_t)ML * DMODEL; // [B,H,L,64]
    float* v_out    = k_out + HLD;                    // [B,H,L,64]

    // 1. casts to bf16
    cast_f32_bf16<<<(ML * DMODEL + 255) / 256, 256, 0, stream>>>(x, xb, ML * DMODEL);
    cast_f32_bf16<<<(NQ * DMODEL + 255) / 256, 256, 0, stream>>>(Wqkv_w, wqkvb, NQ * DMODEL);
    cast_f32_bf16<<<(DMODEL * DMODEL + 255) / 256, 256, 0, stream>>>(Wout_w, woutb, DMODEL * DMODEL);

    // 2. QKV projection: [4096,1024] x [3072,1024]^T
    gemm_bf16_nt<<<dim3(NQ / BN, ML / BM), 256, 0, stream>>>(xb, wqkvb, Wqkv_b, qkv,
                                                             ML, NQ, DMODEL);

    // 3. RoPE + head split (+ emit f32 k,v outputs)
    rope_split<<<(BATCH * NHEAD * SEQ * 32) / 256, 256, 0, stream>>>(qkv, qb, kb, vb,
                                                                     k_out, v_out);

    // 4. causal flash attention
    attn_fa<<<dim3(BATCH * NHEAD, SEQ / 128), 256, 0, stream>>>(qb, kb, vb, attn);

    // 5. output projection
    cast_f32_bf16<<<(ML * DMODEL + 255) / 256, 256, 0, stream>>>(attn, attnb, ML * DMODEL);
    gemm_bf16_nt<<<dim3(DMODEL / BN, ML / BM), 256, 0, stream>>>(attnb, woutb, Wout_b,
                                                                 out_main, ML, DMODEL, DMODEL);
}
